// A_63376537419954
// MI455X (gfx1250) — compile-verified, matrix operand order checked by reading, not run on hardware
//
#include <hip/hip_runtime.h>

// Problem sizes (fixed by the reference).
#define NN     16384      // nodes / sequence length
#define EE     120        // feature dim E
#define E2     240
#define E4     480
#define GG     128        // graphs
#define NEDGE  (NN * 16)  // 262144 edges
#define BN_EPS 1e-5f
#define LN_EPS 1e-5f

// ---------------- WMMA types (CDNA5, wave32) ----------------
typedef __attribute__((ext_vector_type(16))) __bf16 v16bf;
typedef __attribute__((ext_vector_type(8)))  float  v8f;
typedef __attribute__((ext_vector_type(4)))  unsigned int u32x4;

union FragBF { u32x4 q[2]; v16bf v; };

__device__ __forceinline__ unsigned short f2bf(float f) {
  unsigned int u = __float_as_uint(f);
  u += 0x7FFFu + ((u >> 16) & 1u);          // round-to-nearest-even
  return (unsigned short)(u >> 16);
}
__device__ __forceinline__ float sigm(float x) { return 1.0f / (1.0f + __expf(-x)); }

// ---------------- small utility kernels ----------------
__global__ void ak_fill0(float* p, int n) {
  int i = blockIdx.x * blockDim.x + threadIdx.x;
  if (i < n) p[i] = 0.0f;
}
__global__ void ak_vadd(const float* __restrict__ a, const float* __restrict__ b,
                        float* __restrict__ o, int n) {
  int i = blockIdx.x * blockDim.x + threadIdx.x;
  if (i < n) o[i] = a[i] + b[i];
}
__global__ void ak_transpose(const float* __restrict__ X, float* __restrict__ Y,
                             int rows, int cols) {
  int i = blockIdx.x * blockDim.x + threadIdx.x;
  if (i >= rows * cols) return;
  int r = i / cols, c = i % cols;
  Y[(size_t)c * rows + r] = X[i];
}
// f32 (rows x cols) -> bf16 (prows x pcols), zero padded
__global__ void ak_cast_pad(const float* __restrict__ X, unsigned short* __restrict__ Y,
                            int rows, int cols, int prows, int pcols) {
  int i = blockIdx.x * blockDim.x + threadIdx.x;
  if (i >= prows * pcols) return;
  int r = i / pcols, c = i % pcols;
  float v = (r < rows && c < cols) ? X[(size_t)r * cols + c] : 0.0f;
  Y[i] = f2bf(v);
}
// im2col for k=3, pad=1 conv: A[t, c*3+k] = X[c, t+k-1]; X addressed via strides
__global__ void ak_im2col3(const float* __restrict__ X, unsigned short* __restrict__ A,
                           int T, int C, long long sc, long long st, int Kp) {
  int i = blockIdx.x * blockDim.x + threadIdx.x;
  if (i >= T * Kp) return;
  int t = i / Kp, col = i % Kp;
  float v = 0.0f;
  if (col < C * 3) {
    int c = col / 3, k = col % 3;
    int tt = t + k - 1;
    if (tt >= 0 && tt < T) v = X[(long long)c * sc + (long long)tt * st];
  }
  A[i] = f2bf(v);
}

// ---------------- WMMA bf16 GEMM: C(MxNreal) = A(MxKp) @ W(NpadxKp)^T + bias ----------------
// One wave computes one 16x16 tile. Fragment layout per CDNA5 ISA 7.12.2:
// lane half=lane>>4 selects k in {half*8..half*8+7} and {16+half*8..+7} -> two b128 loads.
__global__ __launch_bounds__(128)
void ak_gemm_bf16(const unsigned short* __restrict__ A, const unsigned short* __restrict__ W,
                  const float* __restrict__ bias, float* __restrict__ C,
                  int M, int Nreal, int Npad, int Kp, int act) {
  const int lane  = threadIdx.x & 31;
  const int wave  = threadIdx.x >> 5;
  const int tileM = blockIdx.x;
  const int tileN = blockIdx.y * 4 + wave;
  if (tileN * 16 >= Npad) return;                 // whole-wave early exit (EXEC stays all-1s)
  const int half  = lane >> 4;
  const int i16   = lane & 15;
  const unsigned short* Arow = A + (size_t)(tileM * 16 + i16) * Kp;
  const unsigned short* Wrow = W + (size_t)(tileN * 16 + i16) * Kp;
  v8f acc = {};
  for (int k0 = 0; k0 < Kp; k0 += 32) {
    FragBF a, b;
    a.q[0] = *(const u32x4*)(Arow + k0 + half * 8);
    a.q[1] = *(const u32x4*)(Arow + k0 + 16 + half * 8);
    b.q[0] = *(const u32x4*)(Wrow + k0 + half * 8);
    b.q[1] = *(const u32x4*)(Wrow + k0 + 16 + half * 8);
    acc = __builtin_amdgcn_wmma_f32_16x16x32_bf16(false, a.v, false, b.v,
                                                  (short)0, acc, false, false);
  }
  const int col = tileN * 16 + i16;               // D: col = lane&15, row = r + 8*half
  if (col < Nreal) {
    float bv = bias ? bias[col] : 0.0f;
    for (int r = 0; r < 8; ++r) {
      float v = acc[r] + bv;
      if (act == 1) v = fmaxf(v, 0.0f);
      C[(size_t)(tileM * 16 + r + 8 * half) * Nreal + col] = v;
    }
  }
}

// ---------------- persistent LSTM recurrence (one WG per direction) ----------------
// gates g[j] = xg[t,j] + sum_d WhhT[d,j]*h[d]; j<480 one thread each; order i,f,g,o.
__global__ __launch_bounds__(512)
void ak_lstm_dir(const float* __restrict__ xgF, const float* __restrict__ xgB,
                 const float* __restrict__ whTF, const float* __restrict__ whTB,
                 float* __restrict__ Hout, int T, int ldH) {
  const int dir = blockIdx.x;                     // 0 = forward, 1 = backward
  const float* xg  = dir ? xgB : xgF;
  const float* whT = dir ? whTB : whTF;           // (120 x 480), coalesced over j
  __shared__ float hS[EE];
  __shared__ float cS[EE];
  __shared__ float gS[E4];
  const int j = threadIdx.x;
  if (j < EE) { hS[j] = 0.0f; cS[j] = 0.0f; }
  __syncthreads();
  for (int s = 0; s < T; ++s) {
    const int t = dir ? (T - 1 - s) : s;
    if (j < E4) {
      float acc = xg[(size_t)t * E4 + j];
      #pragma unroll 8
      for (int d = 0; d < EE; ++d)
        acc = fmaf(whT[d * E4 + j], hS[d], acc);  // hS[d] is an LDS broadcast
      gS[j] = acc;
    }
    __syncthreads();
    if (j < EE) {
      float ig = sigm(gS[j]);
      float fg = sigm(gS[j + EE]);
      float gg = tanhf(gS[j + 2 * EE]);
      float og = sigm(gS[j + 3 * EE]);
      float c  = fg * cS[j] + ig * gg;
      float h  = og * tanhf(c);
      cS[j] = c; hS[j] = h;
      Hout[(size_t)t * ldH + dir * EE + j] = h;
    }
    __syncthreads();
  }
}

__global__ void ak_layernorm(const float* __restrict__ X, const float* __restrict__ g,
                             const float* __restrict__ b, float* __restrict__ Y, int N, int D) {
  int n = blockIdx.x * blockDim.x + threadIdx.x;
  if (n >= N) return;
  const float* x = X + (size_t)n * D;
  float mu = 0.0f;
  for (int i = 0; i < D; ++i) mu += x[i];
  mu /= (float)D;
  float var = 0.0f;
  for (int i = 0; i < D; ++i) { float d = x[i] - mu; var += d * d; }
  var /= (float)D;
  float inv = rsqrtf(var + LN_EPS);
  float* y = Y + (size_t)n * D;
  for (int i = 0; i < D; ++i) y[i] = (x[i] - mu) * inv * g[i] + b[i];
}

// O[n,c] = relu(X[n,c] + G2[n,c]) + V[n,c]; X addressed via strides (node- or channel-major)
__global__ void ak_resadd(const float* __restrict__ Xin, long long xs_n, long long xs_c,
                          const float* __restrict__ G2, const float* __restrict__ Vadd,
                          float* __restrict__ O, int N, int C) {
  int i = blockIdx.x * blockDim.x + threadIdx.x;
  if (i >= N * C) return;
  int n = i / C, c = i % C;
  float x = Xin[(long long)n * xs_n + (long long)c * xs_c];
  O[i] = fmaxf(x + G2[i], 0.0f) + Vadd[i];
}

// ---------------- dilated conv block (small-channel, channel-major) ----------------
__global__ void ak_conv1x1_bn_prelu(const float* __restrict__ X, const float* __restrict__ w,
    const float* __restrict__ b, const float* __restrict__ bg, const float* __restrict__ bb,
    const float* __restrict__ bm, const float* __restrict__ bv, const float* __restrict__ aPtr,
    float* __restrict__ O, int T, int Cin, int Cout) {
  int i = blockIdx.x * blockDim.x + threadIdx.x;
  if (i >= Cout * T) return;
  int j = i / T, t = i % T;
  float acc = b[j];
  for (int c = 0; c < Cin; ++c) acc = fmaf(w[j * Cin + c], X[(size_t)c * T + t], acc);
  float sc = bg[j] * rsqrtf(bv[j] + BN_EPS);
  float y  = (acc - bm[j]) * sc + bb[j];
  float a  = *aPtr;
  O[i] = (y >= 0.0f) ? y : a * y;
}
__global__ void ak_dconv3(const float* __restrict__ X, const float* __restrict__ w,
                          float* __restrict__ O, int T, int Cin, int Cout, int dil) {
  int i = blockIdx.x * blockDim.x + threadIdx.x;
  if (i >= Cout * T) return;
  int j = i / T, t = i % T;
  float acc = 0.0f;
  for (int c = 0; c < Cin; ++c) {
    const float* wr = w + (size_t)(j * Cin + c) * 3;
    int tm = t - dil, tp = t + dil;
    float xm = (tm >= 0) ? X[(size_t)c * T + tm] : 0.0f;
    float x0 = X[(size_t)c * T + t];
    float xp = (tp < T) ? X[(size_t)c * T + tp] : 0.0f;
    acc = fmaf(wr[0], xm, fmaf(wr[1], x0, fmaf(wr[2], xp, acc)));
  }
  O[i] = acc;
}
__global__ void ak_dcb_combine(const float* __restrict__ d1, const float* __restrict__ d2,
    const float* __restrict__ d4, const float* __restrict__ d8, const float* __restrict__ d16,
    const float* __restrict__ bg, const float* __restrict__ bb, const float* __restrict__ bm,
    const float* __restrict__ bv, const float* __restrict__ aPtr, float* __restrict__ O,
    int T, int n1, int n, int Cout) {
  int i = blockIdx.x * blockDim.x + threadIdx.x;
  if (i >= Cout * T) return;
  int c = i / T, t = i % T;
  float v;
  if (c < n1) v = d1[(size_t)c * T + t];
  else {
    int blk = (c - n1) / n, cc = (c - n1) % n;
    v = d2[(size_t)cc * T + t];
    if (blk >= 1) v += d4[(size_t)cc * T + t];
    if (blk >= 2) v += d8[(size_t)cc * T + t];
    if (blk >= 3) v += d16[(size_t)cc * T + t];
  }
  float sc = bg[c] * rsqrtf(bv[c] + BN_EPS);
  float y  = (v - bm[c]) * sc + bb[c];
  float a  = *aPtr;
  O[i] = (y >= 0.0f) ? y : a * y;
}

// ---------------- segment / graph ops ----------------
__global__ void ak_seg_accum(const float* __restrict__ X, const int* __restrict__ seg,
                             float* __restrict__ sum, float* __restrict__ cnt, int N, int C) {
  int i = blockIdx.x * blockDim.x + threadIdx.x;
  if (i >= N * C) return;
  int n = i / C, c = i % C;
  atomicAdd(&sum[(size_t)seg[n] * C + c], X[i]);
  if (c == 0) atomicAdd(&cnt[seg[n]], 1.0f);
}
__global__ void ak_seg_scale(float* __restrict__ sum, const float* __restrict__ cnt, int G, int C) {
  int i = blockIdx.x * blockDim.x + threadIdx.x;
  if (i >= G * C) return;
  float k = cnt[i / C];
  sum[i] = (k > 0.0f) ? sum[i] / k : 0.0f;
}
__global__ void ak_edge_scatter(const float* __restrict__ H, const int* __restrict__ src,
                                const int* __restrict__ dst, float* __restrict__ agg,
                                int NE, int C) {
  int e = blockIdx.x, c = threadIdx.x;
  if (e >= NE || c >= C) return;
  atomicAdd(&agg[(size_t)dst[e] * C + c], H[(size_t)src[e] * C + c]);
}
// out = gin + scN * tanh(Z1[seg]) + v1 * tanh(Z2[seg])   (m2s degenerates to per-graph Z)
__global__ void ak_final(const float* __restrict__ gin, const float* __restrict__ scN,
                         const float* __restrict__ v1, const float* __restrict__ Z1,
                         const float* __restrict__ Z2, const int* __restrict__ seg,
                         float* __restrict__ out, int N, int C) {
  int i = blockIdx.x * blockDim.x + threadIdx.x;
  if (i >= N * C) return;
  int n = i / C, c = i % C;
  int g = seg[n];
  out[i] = gin[i] + scN[i] * tanhf(Z1[(size_t)g * C + c]) + v1[i] * tanhf(Z2[(size_t)g * C + c]);
}

// ============================================================================
extern "C" void kernel_launch(void* const* d_in, const int* in_sizes, int n_in,
                              void* d_out, int out_size, void* d_ws, size_t ws_size,
                              hipStream_t stream) {
  (void)in_sizes; (void)n_in; (void)out_size; (void)ws_size;
  // -------- inputs (setup_inputs() insertion order, DFS through nested dicts) --------
  const float* v      = (const float*)d_in[0];
  const float* data_x = (const float*)d_in[2];
  const int*   seg    = (const int*)d_in[4];
  const int*   e_src  = (const int*)d_in[5];
  const int*   e_dst  = e_src + NEDGE;
  const float *wih0f = (const float*)d_in[6],  *whh0f = (const float*)d_in[7];
  const float *bih0f = (const float*)d_in[8],  *bhh0f = (const float*)d_in[9];
  const float *wih0b = (const float*)d_in[10], *whh0b = (const float*)d_in[11];
  const float *bih0b = (const float*)d_in[12], *bhh0b = (const float*)d_in[13];
  const float *wih1f = (const float*)d_in[14], *whh1f = (const float*)d_in[15];
  const float *bih1f = (const float*)d_in[16], *bhh1f = (const float*)d_in[17];
  const float *wih1b = (const float*)d_in[18], *whh1b = (const float*)d_in[19];
  const float *bih1b = (const float*)d_in[20], *bhh1b = (const float*)d_in[21];
  const float *lng = (const float*)d_in[22], *lnb = (const float*)d_in[23];
  const float *fcw = (const float*)d_in[24], *fcb = (const float*)d_in[25];
  const float *rw1 = (const float*)d_in[26], *rb1 = (const float*)d_in[27];
  const float *rw2 = (const float*)d_in[28], *rb2 = (const float*)d_in[29];
  // conv_seq: 3 blocks x 17 leaves starting at 30 (helper.WB/bB and edge_weight are dead)
  const float *WA  = (const float*)d_in[81], *bA  = (const float*)d_in[82];
  const float *Bw  = (const float*)d_in[85], *Bb  = (const float*)d_in[86];
  const float *gw1 = (const float*)d_in[87], *gb1 = (const float*)d_in[88];
  const float *gw2 = (const float*)d_in[89], *gb2 = (const float*)d_in[90];
  float* out = (float*)d_out;

  // -------- workspace bump allocator (total ~206 MB) --------
  char* wsBase = (char*)d_ws;
  size_t wsOff = 0;
  auto alloc = [&](size_t bytes) -> void* {
    void* p = wsBase + wsOff;
    wsOff = (wsOff + bytes + 255) & ~(size_t)255;
    return p;
  };
  float* xgF  = (float*)alloc((size_t)NN * E4 * 4);
  float* xgB  = (float*)alloc((size_t)NN * E4 * 4);
  float* h0   = (float*)alloc((size_t)NN * E2 * 4);   // also reused as LN output
  float* h1   = (float*)alloc((size_t)NN * E2 * 4);
  float* x1   = (float*)alloc((size_t)NN * EE * 4);
  float* g1b  = (float*)alloc((size_t)NN * EE * 4);
  float* g2b  = (float*)alloc((size_t)NN * EE * 4);
  float* v1b  = (float*)alloc((size_t)NN * EE * 4);
  float* scN  = (float*)alloc((size_t)NN * EE * 4);
  float* cmA  = (float*)alloc((size_t)EE * NN * 4);
  float* cmB  = (float*)alloc((size_t)EE * NN * 4);
  float* o1   = (float*)alloc((size_t)24 * NN * 4);
  float* dd1  = (float*)alloc((size_t)24 * NN * 4);
  float* dd2  = (float*)alloc((size_t)24 * NN * 4);
  float* dd4  = (float*)alloc((size_t)24 * NN * 4);
  float* dd8  = (float*)alloc((size_t)24 * NN * 4);
  float* dd16 = (float*)alloc((size_t)24 * NN * 4);
  float* ginH = (float*)alloc((size_t)NN * 128 * 4);
  float* ginT = (float*)alloc((size_t)NN * 128 * 4);
  float* aggB = (float*)alloc((size_t)NN * 128 * 4);
  float* ginO = (float*)alloc((size_t)NN * EE * 4);
  float* vbar = (float*)alloc((size_t)GG * EE * 4);
  float* cnt  = (float*)alloc((size_t)GG * 4);
  float* dAb  = (float*)alloc((size_t)GG * 960 * 4);
  float* Z1   = (float*)alloc((size_t)GG * EE * 4);
  float* Z2   = (float*)alloc((size_t)GG * EE * 4);
  float* whT0f = (float*)alloc((size_t)EE * E4 * 4);
  float* whT0b = (float*)alloc((size_t)EE * E4 * 4);
  float* whT1f = (float*)alloc((size_t)EE * E4 * 4);
  float* whT1b = (float*)alloc((size_t)EE * E4 * 4);
  float* bs0f = (float*)alloc(E4 * 4);
  float* bs0b = (float*)alloc(E4 * 4);
  float* bs1f = (float*)alloc(E4 * 4);
  float* bs1b = (float*)alloc(E4 * 4);
  unsigned short* bfA    = (unsigned short*)alloc((size_t)NN * 384 * 2);  // shared A-staging
  unsigned short* w0f_bf = (unsigned short*)alloc((size_t)E4 * 128 * 2);
  unsigned short* w0b_bf = (unsigned short*)alloc((size_t)E4 * 128 * 2);
  unsigned short* w1f_bf = (unsigned short*)alloc((size_t)E4 * 256 * 2);
  unsigned short* w1b_bf = (unsigned short*)alloc((size_t)E4 * 256 * 2);
  unsigned short* fcw_bf = (unsigned short*)alloc((size_t)128 * 256 * 2);
  unsigned short* rw1_bf = (unsigned short*)alloc((size_t)128 * 384 * 2);
  unsigned short* rw2_bf = (unsigned short*)alloc((size_t)128 * 384 * 2);
  unsigned short* WA_bf  = (unsigned short*)alloc((size_t)960 * 128 * 2);
  unsigned short* Bw_bf  = (unsigned short*)alloc((size_t)128 * 960 * 2);
  unsigned short* gw1_bf = (unsigned short*)alloc((size_t)128 * 128 * 2);
  unsigned short* gw2_bf = (unsigned short*)alloc((size_t)128 * 128 * 2);

  auto gb = [&](long total) { return dim3((unsigned)((total + 255) / 256)); };
  auto gemm = [&](const unsigned short* A, const unsigned short* W, const float* bias,
                  float* C, int M, int Nreal, int Npad, int Kp, int act) {
    dim3 grid(M / 16, (Npad / 16 + 3) / 4);
    ak_gemm_bf16<<<grid, 128, 0, stream>>>(A, W, bias, C, M, Nreal, Npad, Kp, act);
  };
  auto castpad = [&](const float* X, unsigned short* Y, int r, int c, int pr, int pc) {
    ak_cast_pad<<<gb((long)pr * pc), 256, 0, stream>>>(X, Y, r, c, pr, pc);
  };

  // -------- phase 0: weight prep --------
  ak_vadd<<<gb(E4), 256, 0, stream>>>(bih0f, bhh0f, bs0f, E4);
  ak_vadd<<<gb(E4), 256, 0, stream>>>(bih0b, bhh0b, bs0b, E4);
  ak_vadd<<<gb(E4), 256, 0, stream>>>(bih1f, bhh1f, bs1f, E4);
  ak_vadd<<<gb(E4), 256, 0, stream>>>(bih1b, bhh1b, bs1b, E4);
  ak_transpose<<<gb((long)E4 * EE), 256, 0, stream>>>(whh0f, whT0f, E4, EE);
  ak_transpose<<<gb((long)E4 * EE), 256, 0, stream>>>(whh0b, whT0b, E4, EE);
  ak_transpose<<<gb((long)E4 * EE), 256, 0, stream>>>(whh1f, whT1f, E4, EE);
  ak_transpose<<<gb((long)E4 * EE), 256, 0, stream>>>(whh1b, whT1b, E4, EE);
  castpad(wih0f, w0f_bf, E4, EE, E4, 128);
  castpad(wih0b, w0b_bf, E4, EE, E4, 128);
  castpad(wih1f, w1f_bf, E4, E2, E4, 256);
  castpad(wih1b, w1b_bf, E4, E2, E4, 256);
  castpad(fcw, fcw_bf, EE, E2, 128, 256);
  castpad(rw1, rw1_bf, EE, 360, 128, 384);
  castpad(rw2, rw2_bf, EE, 360, 128, 384);
  castpad(WA, WA_bf, 960, EE, 960, 128);      // WA flat is exactly (960,120)
  castpad(Bw, Bw_bf, EE, 960, 128, 960);
  castpad(gw1, gw1_bf, 128, EE, 128, 128);
  castpad(gw2, gw2_bf, EE, 128, 128, 128);

  // -------- phase 1: LSTM (input GEMMs on WMMA + persistent recurrences) --------
  castpad(v, bfA, NN, EE, NN, 128);
  gemm(bfA, w0f_bf, bs0f, xgF, NN, E4, E4, 128, 0);
  gemm(bfA, w0b_bf, bs0b, xgB, NN, E4, E4, 128, 0);
  ak_lstm_dir<<<dim3(2), 512, 0, stream>>>(xgF, xgB, whT0f, whT0b, h0, NN, E2);
  castpad(h0, bfA, NN, E2, NN, 256);
  gemm(bfA, w1f_bf, bs1f, xgF, NN, E4, E4, 256, 0);
  gemm(bfA, w1b_bf, bs1b, xgB, NN, E4, E4, 256, 0);
  ak_lstm_dir<<<dim3(2), 512, 0, stream>>>(xgF, xgB, whT1f, whT1b, h1, NN, E2);

  // -------- phase 2: LayerNorm + FC (h0 buffer reused as LN output) --------
  ak_layernorm<<<gb(NN), 256, 0, stream>>>(h1, lng, lnb, h0, NN, E2);
  castpad(h0, bfA, NN, E2, NN, 256);
  gemm(bfA, fcw_bf, fcb, x1, NN, EE, 128, 256, 0);

  // -------- phase 3: res block on x1 (node-major), v1 = res(x1)+v --------
  ak_im2col3<<<gb((long)NN * 384), 256, 0, stream>>>(x1, bfA, NN, EE, 1LL, (long long)EE, 384);
  gemm(bfA, rw1_bf, rb1, g1b, NN, EE, 128, 384, 1);
  ak_im2col3<<<gb((long)NN * 384), 256, 0, stream>>>(g1b, bfA, NN, EE, 1LL, (long long)EE, 384);
  gemm(bfA, rw2_bf, rb2, g2b, NN, EE, 128, 384, 0);
  ak_resadd<<<gb((long)NN * EE), 256, 0, stream>>>(x1, (long long)EE, 1LL, g2b, v, v1b, NN, EE);

  // -------- phase 4: dilated conv sequence on y = v^T (channel-major) --------
  ak_transpose<<<gb((long)NN * EE), 256, 0, stream>>>(v, cmA, NN, EE);
  struct Dcb { int cin, cout, n, n1, base; };
  const Dcb dcbs[3] = { {120, 32, 6, 8, 30}, {32, 64, 12, 16, 47}, {64, 120, 24, 24, 64} };
  float* cur = cmA; float* nxt = cmB;
  for (int bi = 0; bi < 3; ++bi) {
    const Dcb& d = dcbs[bi];
    const float* c1w = (const float*)d_in[d.base + 0];
    const float* c1b = (const float*)d_in[d.base + 1];
    const float* b1g = (const float*)d_in[d.base + 2];
    const float* b1b = (const float*)d_in[d.base + 3];
    const float* b1m = (const float*)d_in[d.base + 4];
    const float* b1v = (const float*)d_in[d.base + 5];
    const float* p1  = (const float*)d_in[d.base + 6];
    const float* d1w = (const float*)d_in[d.base + 7];
    const float* d2w = (const float*)d_in[d.base + 8];
    const float* d4w = (const float*)d_in[d.base + 9];
    const float* d8w = (const float*)d_in[d.base + 10];
    const float* d16w = (const float*)d_in[d.base + 11];
    const float* b2g = (const float*)d_in[d.base + 12];
    const float* b2b = (const float*)d_in[d.base + 13];
    const float* b2m = (const float*)d_in[d.base + 14];
    const float* b2v = (const float*)d_in[d.base + 15];
    const float* p2  = (const float*)d_in[d.base + 16];
    ak_conv1x1_bn_prelu<<<gb((long)d.n * NN), 256, 0, stream>>>(
        cur, c1w, c1b, b1g, b1b, b1m, b1v, p1, o1, NN, d.cin, d.n);
    ak_dconv3<<<gb((long)d.n1 * NN), 256, 0, stream>>>(o1, d1w, dd1, NN, d.n, d.n1, 1);
    ak_dconv3<<<gb((long)d.n * NN), 256, 0, stream>>>(o1, d2w, dd2, NN, d.n, d.n, 2);
    ak_dconv3<<<gb((long)d.n * NN), 256, 0, stream>>>(o1, d4w, dd4, NN, d.n, d.n, 4);
    ak_dconv3<<<gb((long)d.n * NN), 256, 0, stream>>>(o1, d8w, dd8, NN, d.n, d.n, 8);
    ak_dconv3<<<gb((long)d.n * NN), 256, 0, stream>>>(o1, d16w, dd16, NN, d.n, d.n, 16);
    ak_dcb_combine<<<gb((long)d.cout * NN), 256, 0, stream>>>(
        dd1, dd2, dd4, dd8, dd16, b2g, b2b, b2m, b2v, p2, nxt, NN, d.n1, d.n, d.cout);
    float* tmp = cur; cur = nxt; nxt = tmp;
  }
  float* sc_cm = cur;  // (120 x 16384) channel-major

  // -------- phase 5: res block on sc (channel-major input), scN = res(sc)+v --------
  ak_im2col3<<<gb((long)NN * 384), 256, 0, stream>>>(sc_cm, bfA, NN, EE, (long long)NN, 1LL, 384);
  gemm(bfA, rw1_bf, rb1, g1b, NN, EE, 128, 384, 1);
  ak_im2col3<<<gb((long)NN * 384), 256, 0, stream>>>(g1b, bfA, NN, EE, 1LL, (long long)EE, 384);
  gemm(bfA, rw2_bf, rb2, g2b, NN, EE, 128, 384, 0);
  ak_resadd<<<gb((long)NN * EE), 256, 0, stream>>>(sc_cm, 1LL, (long long)NN, g2b, v, scN, NN, EE);

  // -------- phase 6: m2s (degenerate softmax -> segment means + tiny GEMMs) --------
  auto m2s = [&](const float* Xnodes, float* Zout) {
    ak_fill0<<<gb((long)GG * EE), 256, 0, stream>>>(vbar, GG * EE);
    ak_fill0<<<gb(GG), 256, 0, stream>>>(cnt, GG);
    ak_seg_accum<<<gb((long)NN * EE), 256, 0, stream>>>(Xnodes, seg, vbar, cnt, NN, EE);
    ak_seg_scale<<<gb((long)GG * EE), 256, 0, stream>>>(vbar, cnt, GG, EE);
    castpad(vbar, bfA, GG, EE, GG, 128);
    gemm(bfA, WA_bf, bA, dAb, GG, 960, 960, 128, 0);
    castpad(dAb, bfA, GG, 960, GG, 960);
    gemm(bfA, Bw_bf, Bb, Zout, GG, EE, 128, 960, 0);     // tanh fused into final combine
  };
  m2s(x1, Z1);
  m2s(scN, Z2);

  // -------- phase 7: GIN --------
  ak_fill0<<<gb((long)NN * EE), 256, 0, stream>>>(aggB, NN * EE);
  ak_edge_scatter<<<dim3(NEDGE), 128, 0, stream>>>(data_x, e_src, e_dst, aggB, NEDGE, EE);
  ak_vadd<<<gb((long)NN * EE), 256, 0, stream>>>(data_x, aggB, ginT, NN * EE);
  castpad(ginT, bfA, NN, EE, NN, 128);
  gemm(bfA, gw1_bf, gb1, ginH, NN, 128, 128, 128, 1);
  ak_fill0<<<gb((long)NN * 128), 256, 0, stream>>>(aggB, NN * 128);
  ak_edge_scatter<<<dim3(NEDGE), 128, 0, stream>>>(ginH, e_src, e_dst, aggB, NEDGE, 128);
  ak_vadd<<<gb((long)NN * 128), 256, 0, stream>>>(ginH, aggB, ginT, NN * 128);
  castpad(ginT, bfA, NN, 128, NN, 128);
  gemm(bfA, gw2_bf, gb2, ginO, NN, EE, 128, 128, 1);

  // -------- phase 8: final combine --------
  ak_final<<<gb((long)NN * EE), 256, 0, stream>>>(ginO, scN, v1b, Z1, Z2, seg, out, NN, EE);
}
